// TgGIN_7189775253562
// MI455X (gfx1250) — matrix operand order repaired and checked by reading, not compiled, not run on hardware
//
#include <hip/hip_runtime.h>

typedef __attribute__((ext_vector_type(16))) _Float16 v16h;
typedef __attribute__((ext_vector_type(8)))  _Float16 v8h;
typedef __attribute__((ext_vector_type(8)))  float    v8f;

// ---------------------------------------------------------------------------
// GEMM: Out[r, 0..63] = (A0[r,:] (+ A1[r,:])) @ W^T + bias   (optionally ReLU)
// A is [nRows, KDIM] f32, W is [64, KDIM] f32 row-major, Out is [nRows, 64].
// Block = 128 threads = 4 waves (wave32). Each block covers 64 rows; each
// wave computes a 16x64 slab via 4 accumulators of v_wmma_f32_16x16x32_f16.
// ---------------------------------------------------------------------------
template <int KDIM, bool FUSE_ADD, bool RELU>
__global__ __launch_bounds__(128)
void gin_gemm_wmma(const float* __restrict__ A0,
                   const float* __restrict__ A1,
                   const float* __restrict__ W,
                   const float* __restrict__ bias,
                   float* __restrict__ Out,
                   int nRows)
{
  __shared__ __attribute__((aligned(32))) _Float16 sA[64 * KDIM];
  __shared__ __attribute__((aligned(32))) _Float16 sW[64 * KDIM];

  const int tid     = threadIdx.x;     // 0..127
  const int rowBase = blockIdx.x * 64;

  // Stage weights (f32 -> f16) into LDS: [64][KDIM] row-major.
  for (int i = tid; i < 64 * KDIM; i += 128)
    sW[i] = (_Float16)W[i];

  // Stage A tile (f32 -> f16), fusing the GIN "x + agg" add when requested.
  // Out-of-range rows become zeros so WMMA runs with full EXEC.
  for (int i = tid; i < 64 * KDIM; i += 128) {
    const int r  = i / KDIM;
    const int k  = i - r * KDIM;
    const int gr = rowBase + r;
    float v = 0.0f;
    if (gr < nRows) {
      v = A0[(size_t)gr * KDIM + k];
      if constexpr (FUSE_ADD) v += A1[(size_t)gr * KDIM + k];
    }
    sA[i] = (_Float16)v;
  }
  __syncthreads();

  const int wave   = tid >> 5;   // 4 waves, 16 rows each
  const int lane   = tid & 31;
  const int laneLo = lane & 15;
  const int laneHi = lane >> 4;

  v8f acc[4] = {};               // 4 N-tiles of 16 -> 64 output columns

  // A-fragment row for this lane (both lane halves carry rows M=0..15).
  const int aRow = wave * 16 + laneLo;

  #pragma unroll
  for (int kk = 0; kk < KDIM; kk += 32) {
    // A k order used here (contiguous, same as B): lanes 0-15 hold K [0..16),
    // lanes 16-31 hold K [16..32) of this chunk.
    const _Float16* ap = &sA[aRow * KDIM + kk + laneHi * 16];
    const v8h alo = *(const v8h*)(ap);
    const v8h ahi = *(const v8h*)(ap + 8);
    v16h a;
    #pragma unroll
    for (int j = 0; j < 8; ++j) { a[j] = alo[j]; a[8 + j] = ahi[j]; }

    #pragma unroll
    for (int t = 0; t < 4; ++t) {
      // B (32x16, K x N): lane = column n, contiguous 16 K values per lane
      // half; B[k][n] = W[n][k] so h = A @ W^T.
      const _Float16* bp = &sW[(t * 16 + laneLo) * KDIM + kk + laneHi * 16];
      const v16h b = *(const v16h*)bp;
      acc[t] = __builtin_amdgcn_wmma_f32_16x16x32_f16(
          /*neg_a=*/false, a, /*neg_b=*/false, b,
          /*c_mod=*/(short)0, acc[t], /*reuse_a=*/false, /*reuse_b=*/false);
    }
  }

  // C/D 16x16 f32 layout: lane = column (laneLo), VGPR r = row laneHi*8+r.
  #pragma unroll
  for (int t = 0; t < 4; ++t) {
    const int   n  = t * 16 + laneLo;
    const float bv = bias[n];
    #pragma unroll
    for (int r = 0; r < 8; ++r) {
      const int gr = rowBase + wave * 16 + laneHi * 8 + r;
      if (gr < nRows) {
        float v = acc[t][r] + bv;
        if constexpr (RELU) v = fmaxf(v, 0.0f);
        Out[(size_t)gr * 64 + n] = v;
      }
    }
  }
}

// ---------------------------------------------------------------------------
// Edge scatter-add: Agg[dst[e], :] += H[src[e], :].  16 lanes per edge, each
// lane moves one float4 (coalesced 256B gather per edge), fp32 atomics to L2.
// ---------------------------------------------------------------------------
__global__ __launch_bounds__(256)
void gin_scatter_add(const float* __restrict__ H,
                     const int* __restrict__ src,
                     const int* __restrict__ dst,
                     float* __restrict__ Agg,
                     int nEdges)
{
  const int t     = blockIdx.x * 256 + threadIdx.x;
  const int total = nEdges * 16;
  if (t >= total) return;
  const int e = t >> 4;
  const int c = (t & 15) << 2;         // float offset within the 64-wide row
  const int s = src[e];
  const int d = dst[e];
  const float4 v = *(const float4*)(H + (size_t)s * 64 + c);
  float* p = Agg + (size_t)d * 64 + c;
  __hip_atomic_fetch_add(p + 0, v.x, __ATOMIC_RELAXED, __HIP_MEMORY_SCOPE_AGENT);
  __hip_atomic_fetch_add(p + 1, v.y, __ATOMIC_RELAXED, __HIP_MEMORY_SCOPE_AGENT);
  __hip_atomic_fetch_add(p + 2, v.z, __ATOMIC_RELAXED, __HIP_MEMORY_SCOPE_AGENT);
  __hip_atomic_fetch_add(p + 3, v.w, __ATOMIC_RELAXED, __HIP_MEMORY_SCOPE_AGENT);
}

// ---------------------------------------------------------------------------
// Pipeline:
//   h_pre = x @ W_pre^T + b_pre                              (WMMA, K=128)
//   agg   = segment_sum(h_pre[src], dst)                     (atomics)
//   h1    = relu((h_pre + agg) @ W_first^T + b_first)        (WMMA, K=64)
//   agg   = segment_sum(h1[src], dst)                        (atomics)
//   out   = (h1 + agg) @ W_out^T + b_out                     (WMMA, K=64)
// ---------------------------------------------------------------------------
extern "C" void kernel_launch(void* const* d_in, const int* in_sizes, int n_in,
                              void* d_out, int out_size, void* d_ws, size_t ws_size,
                              hipStream_t stream)
{
  const float* x     = (const float*)d_in[0];
  const int*   ei    = (const int*)  d_in[1];
  const float* W_pre = (const float*)d_in[2];
  const float* b_pre = (const float*)d_in[3];
  const float* W_fst = (const float*)d_in[4];
  const float* b_fst = (const float*)d_in[5];
  const float* Wout  = (const float*)d_in[6];
  const float* bout  = (const float*)d_in[7];

  const int nNodes = in_sizes[0] / 128;   // x is [N, 128]
  const int nEdges = in_sizes[1] / 2;     // edge_index is [2, E]
  const int* src = ei;
  const int* dst = ei + nEdges;

  float* h_pre = (float*)d_ws;
  float* agg   = h_pre + (size_t)nNodes * 64;
  float* h1    = agg   + (size_t)nNodes * 64;

  const size_t featBytes = (size_t)nNodes * 64 * sizeof(float);
  const dim3 gBlk((nNodes + 63) / 64);
  const dim3 sBlk((nEdges * 16 + 255) / 256);

  hipMemsetAsync(agg, 0, featBytes, stream);
  gin_gemm_wmma<128, false, false><<<gBlk, 128, 0, stream>>>(
      x, (const float*)nullptr, W_pre, b_pre, h_pre, nNodes);
  gin_scatter_add<<<sBlk, 256, 0, stream>>>(h_pre, src, dst, agg, nEdges);
  gin_gemm_wmma<64, true, true><<<gBlk, 128, 0, stream>>>(
      h_pre, agg, W_fst, b_fst, h1, nNodes);
  hipMemsetAsync(agg, 0, featBytes, stream);
  gin_scatter_add<<<sBlk, 256, 0, stream>>>(h1, src, dst, agg, nEdges);
  gin_gemm_wmma<64, true, false><<<gBlk, 128, 0, stream>>>(
      h1, agg, Wout, bout, (float*)d_out, nNodes);

  (void)n_in; (void)out_size; (void)ws_size;
}
